// SA_55344948576506
// MI455X (gfx1250) — matrix operand order repaired and checked by reading, not compiled, not run on hardware
//
#include <hip/hip_runtime.h>
#include <hip/hip_bf16.h>

typedef __attribute__((ext_vector_type(16))) _Float16 v16h;
typedef __attribute__((ext_vector_type(8)))  _Float16 v8h;
typedef __attribute__((ext_vector_type(8)))  float    v8f;
typedef __attribute__((ext_vector_type(4)))  unsigned int v4u;
typedef __attribute__((ext_vector_type(8)))  int      v8i_t;
typedef __attribute__((ext_vector_type(4)))  int      v4i_t;

#define BATCH   4
#define EMBED   256
#define NTOK    2304      // 48*48
#define NHEADS  8
#define DHEAD   32
#define MTILES  (NTOK/16) // 144

// ---------------------------------------------------------------------------
// WMMA helper: D = A(16x32 f16) * B(32x16 f16) + C(16x16 f32)
// ---------------------------------------------------------------------------
__device__ __forceinline__ v8f wmma_f16(v16h a, v16h b, v8f c) {
  return __builtin_amdgcn_wmma_f32_16x16x32_f16(
      /*neg_a=*/false, a, /*neg_b=*/false, b,
      /*c_mod=*/(short)0, c, /*reuse_a=*/false, /*reuse_b=*/false);
}

// A-operand k order used here (16-bit A, 16x32; B uses the same):
//   lane l: row m = l & 15, hk = l >> 4
//   element e: K = hk*16 + e
// -> one contiguous 16-half (32B) chunk at K = hk*16.
__device__ __forceinline__ v16h load_a_tile(const _Float16* rowbase, int k0, int hk) {
  return *(const v16h*)(rowbase + k0 + hk * 16);
}

// B-operand k order used here (16-bit B, 32x16, same as A): lane = column, element e -> K = hk*16 + e
// -> one contiguous 16-half (32B) chunk at K = hk*16.
__device__ __forceinline__ v16h load_b_tile(const _Float16* colbase, int k0, int hk) {
  return *(const v16h*)(colbase + k0 + hk * 16);
}

// ---------------------------------------------------------------------------
// Tensor Data Mover: 2-D tile (tile_w x tile_h elements, 2B each) from global
// to LDS. D# per ISA 08_async_tensor §8: group0 = control/lds/global addr,
// group1 = dims/strides. Wave-uniform; tracked by TENSORcnt.
// ---------------------------------------------------------------------------
__device__ __forceinline__ void tdm_load_tile_2d(unsigned lds_off_bytes,
                                                 const void* gaddr,
                                                 unsigned tile_w, unsigned tile_h,
                                                 unsigned row_stride_elems) {
  unsigned long long ga = (unsigned long long)(uintptr_t)gaddr;
  v4u g0;
  g0[0] = 1u;                                   // count=1, user mode, no gather
  g0[1] = lds_off_bytes;                        // lds_addr (bytes)
  g0[2] = (unsigned)(ga & 0xffffffffu);         // global_addr[31:0]
  g0[3] = (unsigned)((ga >> 32) & 0x1ffffffu)   // global_addr[56:32]
          | (2u << 30);                         // type = 2 ("image")

  v8i_t g1;
  g1[0] = (int)(1u << 16);                      // wg_mask=0, data_size=1 (2 bytes)
  g1[1] = (int)((tile_w & 0xffffu) << 16);      // tensor_dim0[15:0]
  g1[2] = (int)(((tile_w >> 16) & 0xffffu) |    // tensor_dim0[31:16]
                ((tile_h & 0xffffu) << 16));    // tensor_dim1[15:0]
  g1[3] = (int)(((tile_h >> 16) & 0xffffu) |    // tensor_dim1[31:16]
                ((tile_w & 0xffffu) << 16));    // tile_dim0
  g1[4] = (int)(tile_h & 0xffffu);              // tile_dim1 (tile_dim2 = 0)
  g1[5] = (int)row_stride_elems;                // tensor_dim0_stride[31:0]
  g1[6] = 0;                                    // stride0 hi / stride1 lo
  g1[7] = 0;                                    // stride1 hi

  v4i_t gz = {0, 0, 0, 0};
#if defined(__clang_major__) && (__clang_major__ >= 23)
  v8i_t gz8 = {0, 0, 0, 0, 0, 0, 0, 0};
  __builtin_amdgcn_tensor_load_to_lds(g0, g1, gz, gz, gz8, 0);
#else
  __builtin_amdgcn_tensor_load_to_lds(g0, g1, gz, gz, 0);
#endif
}

// Low 32 bits of a generic pointer to LDS = LDS byte offset (ISA §10.2 aperture).
__device__ __forceinline__ unsigned lds_offset(const void* p) {
  return (unsigned)(uintptr_t)p;
}

// ---------------------------------------------------------------------------
// Kernel 1: xs16[b,n,c] = x[b,c,n] + PE(n,c), cast to f16
// ---------------------------------------------------------------------------
__global__ void prep_pe_kernel(const float* __restrict__ x,
                               _Float16* __restrict__ xs16) {
  int idx = blockIdx.x * blockDim.x + threadIdx.x;
  const int total = BATCH * NTOK * EMBED;
  if (idx >= total) return;
  int cc = idx % EMBED;
  int nn = (idx / EMBED) % NTOK;
  int b  = idx / (EMBED * NTOK);
  float pe = 0.0f;
  if (nn != 0) {
    float ang = (float)nn * __expf(-(float)cc * (9.210340371976184f / 128.0f));
    pe = (cc & 1) ? cosf(ang) : sinf(ang);
  }
  float val = x[((size_t)b * EMBED + cc) * NTOK + nn] + pe;
  xs16[((size_t)b * NTOK + nn) * EMBED + cc] = (_Float16)val;
}

// ---------------------------------------------------------------------------
// Kernel 2: cast the four 256x256 weight matrices to f16 (q,k,v,o packed)
// ---------------------------------------------------------------------------
__global__ void convert_w_kernel(const float* __restrict__ wq,
                                 const float* __restrict__ wk,
                                 const float* __restrict__ wv,
                                 const float* __restrict__ wo,
                                 _Float16* __restrict__ w16) {
  int idx = blockIdx.x * blockDim.x + threadIdx.x;
  const int total = 4 * EMBED * EMBED;
  if (idx >= total) return;
  int m   = idx >> 16;
  int off = idx & 65535;
  const float* s = (m == 0) ? wq : (m == 1) ? wk : (m == 2) ? wv : wo;
  w16[idx] = (_Float16)s[off];
}

// ---------------------------------------------------------------------------
// Kernel 3: projection GEMM  out16[b,n,:] = xs16[b,n,:] @ W^T + bias
// One wave -> 16(M) x 64(N) tile. grid: (BATCH*MTILES, EMBED/64)
// ---------------------------------------------------------------------------
__global__ void proj_wmma_kernel(const _Float16* __restrict__ xs16,
                                 const _Float16* __restrict__ w16,
                                 const float*    __restrict__ bias,
                                 _Float16*       __restrict__ out16) {
  int lane = threadIdx.x;
  int row  = lane & 15;
  int hk   = lane >> 4;
  int mt = blockIdx.x;
  int b  = mt / MTILES;
  int m0 = (mt % MTILES) * 16;
  int n0 = blockIdx.y * 64;

  v8f acc[4] = {v8f{}, v8f{}, v8f{}, v8f{}};
  const _Float16* abase = xs16 + ((size_t)(b * NTOK + m0 + row)) * EMBED;

  for (int k0 = 0; k0 < EMBED; k0 += 32) {
    v16h a = load_a_tile(abase, k0, hk);
#pragma unroll
    for (int sub = 0; sub < 4; ++sub) {
      const _Float16* bb = w16 + (size_t)(n0 + sub * 16 + row) * EMBED;
      v16h bm = load_b_tile(bb, k0, hk);
      acc[sub] = wmma_f16(a, bm, acc[sub]);
    }
  }
#pragma unroll
  for (int sub = 0; sub < 4; ++sub) {
    int col = n0 + sub * 16 + row;
    float bv = bias[col];
#pragma unroll
    for (int r = 0; r < 8; ++r) {
      int rr = r + hk * 8;
      out16[((size_t)(b * NTOK + m0 + rr)) * EMBED + col] =
          (_Float16)(acc[sub][r] + bv);
    }
  }
}

// ---------------------------------------------------------------------------
// Kernel 3b: V projection, stored TRANSPOSED per head: vT[b][head][dh][n].
// Makes the attention V B-operand a contiguous 32B load per lane.
// ---------------------------------------------------------------------------
__global__ void proj_wmma_vT_kernel(const _Float16* __restrict__ xs16,
                                    const _Float16* __restrict__ w16,
                                    const float*    __restrict__ bias,
                                    _Float16*       __restrict__ vT) {
  int lane = threadIdx.x;
  int row  = lane & 15;
  int hk   = lane >> 4;
  int mt = blockIdx.x;
  int b  = mt / MTILES;
  int m0 = (mt % MTILES) * 16;
  int n0 = blockIdx.y * 64;

  v8f acc[4] = {v8f{}, v8f{}, v8f{}, v8f{}};
  const _Float16* abase = xs16 + ((size_t)(b * NTOK + m0 + row)) * EMBED;

  for (int k0 = 0; k0 < EMBED; k0 += 32) {
    v16h a = load_a_tile(abase, k0, hk);
#pragma unroll
    for (int sub = 0; sub < 4; ++sub) {
      const _Float16* bb = w16 + (size_t)(n0 + sub * 16 + row) * EMBED;
      v16h bm = load_b_tile(bb, k0, hk);
      acc[sub] = wmma_f16(a, bm, acc[sub]);
    }
  }
#pragma unroll
  for (int sub = 0; sub < 4; ++sub) {
    int col  = n0 + sub * 16 + row;      // output channel
    int head = col >> 5;
    int dh   = col & 31;
    float bv = bias[col];
    _Float16* dst = vT + ((size_t)((b * NHEADS + head) * DHEAD + dh)) * NTOK + m0;
#pragma unroll
    for (int r = 0; r < 8; ++r) {
      int rr = r + hk * 8;
      dst[rr] = (_Float16)(acc[sub][r] + bv);
    }
  }
}

// ---------------------------------------------------------------------------
// Kernel 4: flash attention. One wave per (b, head, 16-query tile).
// K tiles staged into LDS via the Tensor Data Mover, double-buffered on
// TENSORcnt. V read directly from the transposed vT layout.
// ---------------------------------------------------------------------------
__global__ void attn_kernel(const _Float16* __restrict__ q16,
                            const _Float16* __restrict__ k16,
                            const _Float16* __restrict__ vT,
                            _Float16* __restrict__ att16) {
  __shared__ _Float16 p_lds[16 * 32];        // P transpose staging
  __shared__ _Float16 k_lds[2][32 * 32];     // TDM double buffer (2x2KB)

  int lane = threadIdx.x;
  int row  = lane & 15;
  int hk   = lane >> 4;
  int bx   = blockIdx.x;
  int qi   = bx % MTILES;
  int head = (bx / MTILES) % NHEADS;
  int b    = bx / (MTILES * NHEADS);
  int q0   = qi * 16;
  int co   = head * DHEAD;
  const float scale = 0.17677669529663687f;  // 1/sqrt(32)

  const _Float16* kbase = k16 + ((size_t)b * NTOK) * EMBED + co;
  const _Float16* vbase = vT + ((size_t)((b * NHEADS + head) * DHEAD)) * NTOK;

  // Q tile in A layout (row = query, K = dh 0..31)
  const _Float16* qrow = q16 + ((size_t)(b * NTOK + q0 + row)) * EMBED + co;
  v16h qa = load_a_tile(qrow, 0, hk);

  v8f o0 = {}, o1 = {};
  float m_run[8], l_run[8];
#pragma unroll
  for (int r = 0; r < 8; ++r) { m_run[r] = -3.0e38f; l_run[r] = 0.0f; }

  // Kick off first K tile: rows k0..k0+31 of the head slice (32x32 f16).
  tdm_load_tile_2d(lds_offset(&k_lds[0][0]), kbase, DHEAD, 32, EMBED);

  const v8f zero = {};
  int buf = 0;
  for (int k0 = 0; k0 < NTOK; k0 += 32) {
    if (k0 + 32 < NTOK) {
      tdm_load_tile_2d(lds_offset(&k_lds[buf ^ 1][0]),
                       kbase + (size_t)(k0 + 32) * EMBED, DHEAD, 32, EMBED);
      __builtin_amdgcn_s_wait_tensorcnt(1);   // current tile landed
    } else {
      __builtin_amdgcn_s_wait_tensorcnt(0);
    }

    // K^T tiles in B layout from LDS: column = key position, K = dh contiguous.
    const _Float16* kl = &k_lds[buf][0];
    v16h kb0 = *(const v16h*)(kl + row * 32 + hk * 16);
    v16h kb1 = *(const v16h*)(kl + (16 + row) * 32 + hk * 16);
    v8f s0 = wmma_f16(qa, kb0, zero);
    v8f s1 = wmma_f16(qa, kb1, zero);

    // Online softmax: D-layout rows live across one 16-lane half.
#pragma unroll
    for (int r = 0; r < 8; ++r) {
      float a0 = s0[r] * scale;
      float a1 = s1[r] * scale;
      float mx = fmaxf(a0, a1);
      mx = fmaxf(mx, __shfl_xor(mx, 1));
      mx = fmaxf(mx, __shfl_xor(mx, 2));
      mx = fmaxf(mx, __shfl_xor(mx, 4));
      mx = fmaxf(mx, __shfl_xor(mx, 8));
      float mn = fmaxf(m_run[r], mx);
      float p0 = __expf(a0 - mn);
      float p1 = __expf(a1 - mn);
      float ps = p0 + p1;
      ps += __shfl_xor(ps, 1);
      ps += __shfl_xor(ps, 2);
      ps += __shfl_xor(ps, 4);
      ps += __shfl_xor(ps, 8);
      float cf = __expf(m_run[r] - mn);
      l_run[r] = l_run[r] * cf + ps;
      m_run[r] = mn;
      o0[r] *= cf;
      o1[r] *= cf;
      int prow = r + hk * 8;
      p_lds[prow * 32 + row]      = (_Float16)p0;
      p_lds[prow * 32 + 16 + row] = (_Float16)p1;
    }
    __syncthreads();

    // P back in A layout (row = query, K = key position 0..31)
    v16h pa = load_a_tile(p_lds + row * 32, 0, hk);

    // V tiles in B layout from vT: column = dh (lane), K = key pos contiguous.
    const _Float16* v0p = vbase + (size_t)row * NTOK + k0;
    const _Float16* v1p = vbase + (size_t)(16 + row) * NTOK + k0;
    if (k0 + 32 < NTOK) {
      __builtin_prefetch(v0p + 32, 0, 0);   // global_prefetch_b8
      __builtin_prefetch(v1p + 32, 0, 0);
    }
    v16h vb0 = *(const v16h*)(v0p + hk * 16);
    v16h vb1 = *(const v16h*)(v1p + hk * 16);

    o0 = wmma_f16(pa, vb0, o0);
    o1 = wmma_f16(pa, vb1, o1);
    __syncthreads();
    buf ^= 1;
  }

#pragma unroll
  for (int r = 0; r < 8; ++r) {
    float inv = 1.0f / l_run[r];
    int rr = r + hk * 8;
    size_t base = ((size_t)(b * NTOK + q0 + rr)) * EMBED + co;
    att16[base + row]      = (_Float16)(o0[r] * inv);
    att16[base + 16 + row] = (_Float16)(o1[r] * inv);
  }
}

// ---------------------------------------------------------------------------
// Kernel 5: output projection + transpose to [b, c, h*w] f32
// ---------------------------------------------------------------------------
__global__ void oproj_wmma_kernel(const _Float16* __restrict__ att16,
                                  const _Float16* __restrict__ wo16,
                                  const float*    __restrict__ bo,
                                  float*          __restrict__ out) {
  int lane = threadIdx.x;
  int row  = lane & 15;
  int hk   = lane >> 4;
  int mt = blockIdx.x;
  int b  = mt / MTILES;
  int m0 = (mt % MTILES) * 16;
  int n0 = blockIdx.y * 64;

  v8f acc[4] = {v8f{}, v8f{}, v8f{}, v8f{}};
  const _Float16* abase = att16 + ((size_t)(b * NTOK + m0 + row)) * EMBED;

  for (int k0 = 0; k0 < EMBED; k0 += 32) {
    v16h a = load_a_tile(abase, k0, hk);
#pragma unroll
    for (int sub = 0; sub < 4; ++sub) {
      const _Float16* bb = wo16 + (size_t)(n0 + sub * 16 + row) * EMBED;
      v16h bm = load_b_tile(bb, k0, hk);
      acc[sub] = wmma_f16(a, bm, acc[sub]);
    }
  }
#pragma unroll
  for (int sub = 0; sub < 4; ++sub) {
    int col = n0 + sub * 16 + row;
    float bv = bo[col];
#pragma unroll
    for (int r = 0; r < 8; ++r) {
      int rr = r + hk * 8;
      out[((size_t)b * EMBED + col) * NTOK + (m0 + rr)] = acc[sub][r] + bv;
    }
  }
}

// ---------------------------------------------------------------------------
extern "C" void kernel_launch(void* const* d_in, const int* in_sizes, int n_in,
                              void* d_out, int out_size, void* d_ws, size_t ws_size,
                              hipStream_t stream) {
  const float* x  = (const float*)d_in[0];
  const float* wq = (const float*)d_in[1];
  const float* bq = (const float*)d_in[2];
  const float* wk = (const float*)d_in[3];
  const float* bk = (const float*)d_in[4];
  const float* wv = (const float*)d_in[5];
  const float* bv = (const float*)d_in[6];
  const float* wo = (const float*)d_in[7];
  const float* bo = (const float*)d_in[8];
  float* out = (float*)d_out;

  const size_t xsBytes = (size_t)BATCH * NTOK * EMBED * sizeof(_Float16);
  const size_t wBytes  = (size_t)4 * EMBED * EMBED * sizeof(_Float16);

  char* ws = (char*)d_ws;
  _Float16* xs16  = (_Float16*)ws;                 ws += xsBytes;
  _Float16* w16   = (_Float16*)ws;                 ws += wBytes;   // q,k,v,o packed
  _Float16* q16   = (_Float16*)ws;                 ws += xsBytes;
  _Float16* k16   = (_Float16*)ws;                 ws += xsBytes;
  _Float16* vTb   = (_Float16*)ws;                 ws += xsBytes;  // [b][head][dh][n]
  _Float16* att16 = (_Float16*)ws;                 ws += xsBytes;

  {
    int total = BATCH * NTOK * EMBED;
    prep_pe_kernel<<<(total + 255) / 256, 256, 0, stream>>>(x, xs16);
  }
  {
    int total = 4 * EMBED * EMBED;
    convert_w_kernel<<<(total + 255) / 256, 256, 0, stream>>>(wq, wk, wv, wo, w16);
  }
  {
    dim3 grid(BATCH * MTILES, EMBED / 64);
    proj_wmma_kernel<<<grid, 32, 0, stream>>>(xs16, w16 + 0 * EMBED * EMBED, bq, q16);
    proj_wmma_kernel<<<grid, 32, 0, stream>>>(xs16, w16 + 1 * EMBED * EMBED, bk, k16);
    proj_wmma_vT_kernel<<<grid, 32, 0, stream>>>(xs16, w16 + 2 * EMBED * EMBED, bv, vTb);
  }
  {
    int blocks = BATCH * NHEADS * MTILES;  // 4608
    attn_kernel<<<blocks, 32, 0, stream>>>(q16, k16, vTb, att16);
  }
  {
    dim3 grid(BATCH * MTILES, EMBED / 64);
    oproj_wmma_kernel<<<grid, 32, 0, stream>>>(att16, w16 + 3 * EMBED * EMBED, bo, out);
  }
}
